// Attention_26560077758997
// MI455X (gfx1250) — compile-verified, matrix operand order checked by reading, not run on hardware
//
#include <hip/hip_runtime.h>

#define S_LEN 2048
#define NH    32
#define NKVH  8
#define HD    128
#define KT    64     // key tile (two 32-key WMMA K-chunks, four 16-key N-tiles)

// (1/sqrt(128)) * log2(e): softmax in exp2 domain so exp lowers to v_exp_f32
#define SCALE_L2E 0.12751741f
#define NEG_BIG   (-1.0e30f)

typedef __attribute__((ext_vector_type(16))) __bf16 v16bf;
typedef __attribute__((ext_vector_type(8)))  __bf16 v8bf;
typedef __attribute__((ext_vector_type(8)))  float  v8f;

// 16-element bf16 fragment: elems 0..7 from p[0..7], 8..15 from p[16..23].
// Matches the CDNA5 16-bit A/B fragment layout (half-wave selects K base).
// Both halves are 16B aligned -> two ds_load_b128.
__device__ __forceinline__ v16bf load_frag16(const __bf16* p) {
  v8bf lo = *(const v8bf*)(p);
  v8bf hi = *(const v8bf*)(p + 16);
  return __builtin_shufflevector(lo, hi, 0, 1, 2, 3, 4, 5, 6, 7,
                                 8, 9, 10, 11, 12, 13, 14, 15);
}

// Convert 8 consecutive floats (two float4 loads) to a packed v8bf.
__device__ __forceinline__ v8bf cvt8_bf16(const float* p) {
  const float4 a = *(const float4*)(p);
  const float4 b = *(const float4*)(p + 4);
  v8bf r;
  r[0] = (__bf16)a.x; r[1] = (__bf16)a.y; r[2] = (__bf16)a.z; r[3] = (__bf16)a.w;
  r[4] = (__bf16)b.x; r[5] = (__bf16)b.y; r[6] = (__bf16)b.z; r[7] = (__bf16)b.w;
  return r;
}

__global__ __launch_bounds__(128, 1)
void attn_fa_bf16(const float* __restrict__ qg, const float* __restrict__ kg,
                  const float* __restrict__ vg, float* __restrict__ og)
{
  // Double-buffered K tile [key][d], rows padded 128->136 (272B = 68 banks:
  // 16 lanes reading 16 rows land on distinct bank quads).
  __shared__ alignas(16) __bf16 ldsK[2][KT][136];
  // Double-buffered V tile pre-transposed [d][key], rows padded 64->72
  // (144B = 36 banks -> conflict-free across 16 lanes).
  __shared__ alignas(16) __bf16 ldsVt[2][HD][72];
  // Per-wave P scratch for C-layout -> A-layout transpose, [m][key], padded.
  __shared__ alignas(16) __bf16 ldsP[4][16][72];

  const int h    = blockIdx.y;          // query head
  const int kvh  = h >> 2;              // GQA: 4 query heads per KV head
  const int q0   = blockIdx.x * 64;     // query block base
  const int tid  = threadIdx.x;
  const int wave = tid >> 5;            // 4 waves, 16 query rows each
  const int lane = tid & 31;
  const int half = lane >> 4;           // half-wave select (ISA fragment layout)
  const int l16  = lane & 15;

  // Cooperative fill mapping: 128 threads cover 64 keys x 128 d (64 floats each)
  const int fkey = tid >> 1;            // key row 0..63
  const int fd0  = (tid & 1) * 64;      // d slice base

  auto fill = [&](int buf, int kb) {
    const float* kp = kg + ((size_t)(kb * KT + fkey) * NKVH + kvh) * HD + fd0;
    const float* vp = vg + ((size_t)(kb * KT + fkey) * NKVH + kvh) * HD + fd0;
    // K: packed 16B LDS stores (8 x ds_store_b128 per thread)
#pragma unroll
    for (int j0 = 0; j0 < 64; j0 += 8)
      *(v8bf*)&ldsK[buf][fkey][fd0 + j0] = cvt8_bf16(kp + j0);
    // V: transposed scatter (scalar b16 stores; buys conflict-free b128
    // fragment loads in the hot P*V loop)
#pragma unroll
    for (int j0 = 0; j0 < 64; j0 += 4) {
      const float4 a = *(const float4*)(vp + j0);
      ldsVt[buf][fd0 + j0 + 0][fkey] = (__bf16)a.x;
      ldsVt[buf][fd0 + j0 + 1][fkey] = (__bf16)a.y;
      ldsVt[buf][fd0 + j0 + 2][fkey] = (__bf16)a.z;
      ldsVt[buf][fd0 + j0 + 3][fkey] = (__bf16)a.w;
    }
  };

  // ---- This wave's 16x128 Q tile once, as 4 bf16 A-fragments (K=32 each) ----
  const int qrow = q0 + wave * 16 + l16;
  const float* qp = qg + ((size_t)qrow * NH + h) * HD;
  v16bf qf[4];
#pragma unroll
  for (int c = 0; c < 4; ++c) {
    const int d0 = c * 32 + half * 8;
    const v8bf lo = cvt8_bf16(qp + d0);
    const v8bf hi = cvt8_bf16(qp + d0 + 16);
    qf[c] = __builtin_shufflevector(lo, hi, 0, 1, 2, 3, 4, 5, 6, 7,
                                    8, 9, 10, 11, 12, 13, 14, 15);
  }

  // Flash-attention running state. C/D layout: lane = N (l16), VGPR v holds
  // M = v + 8*half, so per-lane row stats cover rows (half*8 + 0..7).
  v8f oacc[8] = {};
  float mrow[8], lrow[8];
#pragma unroll
  for (int i = 0; i < 8; ++i) { mrow[i] = NEG_BIG; lrow[i] = 0.0f; }

  const int nkb    = blockIdx.x + 1;    // causal: 64-key tiles covering <= q0+63
  const int maxrow = q0 + wave * 16 + 15;
  const int minrow = q0 + wave * 16;

  fill(0, 0);                            // prime first buffer

  for (int kb = 0; kb < nkb; ++kb) {
    const int kbase = kb * KT;
    const int buf   = kb & 1;

    __syncthreads();                     // tile kb fully staged for all waves

    if (kb + 1 < nkb) {
      fill(buf ^ 1, kb + 1);             // overlap next tile's fill with compute
      if (kb + 2 < nkb) {                // warm L2 two tiles ahead
        const float* kp2 = kg + ((size_t)((kb + 2) * KT + fkey) * NKVH + kvh) * HD + fd0;
        const float* vp2 = vg + ((size_t)((kb + 2) * KT + fkey) * NKVH + kvh) * HD + fd0;
        __builtin_prefetch(kp2, 0, 1);
        __builtin_prefetch(vp2, 0, 1);
      }
    }

    // Wave-uniform causal skip (EXEC stays all-1s inside -> WMMA legal).
    if (kbase <= maxrow) {
      // ---- S = Q K^T : four 16-key N-tiles, K-dim 128 in 4 chunks of 32.
      // sched_barrier after every 2 chains caps in-flight B-fragments so the
      // allocation stays inside the 256-VGPR direct window.
      v8f sc[4];
#pragma unroll
      for (int nt = 0; nt < 4; ++nt) {
        v8f acc = {};
#pragma unroll
        for (int c = 0; c < 4; ++c) {
          v16bf kf = load_frag16(&ldsK[buf][nt * 16 + l16][c * 32 + half * 8]);
          acc = __builtin_amdgcn_wmma_f32_16x16x32_bf16(
              false, qf[c], false, kf, (short)0, acc, false, false);
        }
        sc[nt] = acc;
        if ((nt & 1) == 1) __builtin_amdgcn_sched_barrier(0);
      }

      // ---- online softmax (exp2 domain); reductions across 16-lane halves ----
      const bool needMask = (kbase + KT - 1) > minrow;   // diagonal tile only
#pragma unroll
      for (int r = 0; r < 8; ++r) {
        const int row = q0 + wave * 16 + r + 8 * half;
        float sv[4];
#pragma unroll
        for (int nt = 0; nt < 4; ++nt) {
          sv[nt] = sc[nt][r] * SCALE_L2E;
          if (needMask && (kbase + nt * 16 + l16 > row)) sv[nt] = NEG_BIG;
        }
        float t = fmaxf(fmaxf(sv[0], sv[1]), fmaxf(sv[2], sv[3]));
#pragma unroll
        for (int off = 1; off < 16; off <<= 1) t = fmaxf(t, __shfl_xor(t, off));
        const float mnew = fmaxf(mrow[r], t);
        const float a = exp2f(mrow[r] - mnew);
        float ps = 0.0f;
#pragma unroll
        for (int nt = 0; nt < 4; ++nt) {
          const float p = exp2f(sv[nt] - mnew);
          ps += p;
          sc[nt][r] = p;
        }
#pragma unroll
        for (int off = 1; off < 16; off <<= 1) ps += __shfl_xor(ps, off);
        lrow[r] = lrow[r] * a + ps;
        mrow[r] = mnew;
        // rescale O accumulator row r in place (no alpha[] live range)
#pragma unroll
        for (int t8 = 0; t8 < 8; ++t8) oacc[t8][r] *= a;
      }
      __builtin_amdgcn_sched_barrier(0);

      // ---- transpose P (C layout: lane=key) -> A layout (lane=row) via LDS ----
#pragma unroll
      for (int r = 0; r < 8; ++r) {
        const int m = r + 8 * half;
#pragma unroll
        for (int nt = 0; nt < 4; ++nt)
          ldsP[wave][m][nt * 16 + l16] = (__bf16)sc[nt][r];
      }
      // same-wave LDS ops are in-order; compiler inserts the dscnt waits
      const v16bf pf0 = load_frag16(&ldsP[wave][l16][half * 8]);
      const v16bf pf1 = load_frag16(&ldsP[wave][l16][32 + half * 8]);

      // ---- O += P V : 8 d-tiles of 16, K-dim = 64 keys in 2 chunks of 32.
      // sched_barrier every 2 d-tiles caps in-flight V-fragments.
#pragma unroll
      for (int dt = 0; dt < 8; ++dt) {
        v16bf vf0 = load_frag16(&ldsVt[buf][dt * 16 + l16][half * 8]);
        oacc[dt] = __builtin_amdgcn_wmma_f32_16x16x32_bf16(
            false, pf0, false, vf0, (short)0, oacc[dt], false, false);
        v16bf vf1 = load_frag16(&ldsVt[buf][dt * 16 + l16][32 + half * 8]);
        oacc[dt] = __builtin_amdgcn_wmma_f32_16x16x32_bf16(
            false, pf1, false, vf1, (short)0, oacc[dt], false, false);
        if ((dt & 1) == 1) __builtin_amdgcn_sched_barrier(0);
      }
    }
  }

  // ---- epilogue: O = acc / l, fp32 out [S][H][D] ----
#pragma unroll
  for (int r = 0; r < 8; ++r) {
    const int row = q0 + wave * 16 + r + 8 * half;
    const float inv = 1.0f / lrow[r];
    float* op = og + ((size_t)row * NH + h) * HD;
#pragma unroll
    for (int dt = 0; dt < 8; ++dt)
      op[dt * 16 + l16] = oacc[dt][r] * inv;
  }
}

extern "C" void kernel_launch(void* const* d_in, const int* in_sizes, int n_in,
                              void* d_out, int out_size, void* d_ws, size_t ws_size,
                              hipStream_t stream) {
  const float* q = (const float*)d_in[0];  // [S, H, D]   fp32
  const float* k = (const float*)d_in[1];  // [S, KVH, D] fp32
  const float* v = (const float*)d_in[2];  // [S, KVH, D] fp32
  float* out = (float*)d_out;              // [S, H, D]   fp32

  dim3 grid(S_LEN / 64, NH);               // 32 query blocks x 32 heads
  attn_fa_bf16<<<grid, 128, 0, stream>>>(q, k, v, out);
}
